// SparseMoELayer_46712064311617
// MI455X (gfx1250) — matrix operand order repaired and checked by reading, not compiled, not run on hardware
//
#include <hip/hip_runtime.h>

// ---------------- problem constants ----------------
#define NTOK 8192
#define DMODEL 1024
#define NEXP 8
#define FFN 4096
#define TOPK 2
#define CAP 2560            // int(8192*2*1.25/8)
#define NASSIGN (TOPK * NTOK)   // 16384, slot-major: a = slot*NTOK + n
#define NSLOT (NEXP * CAP)      // 20480
#define KT 64               // K-tile staged in LDS (two 32-wide WMMA substeps)

typedef __bf16 bf16_t;
typedef __attribute__((ext_vector_type(16))) __bf16 v16bf;
typedef __attribute__((ext_vector_type(8)))  __bf16 v8bf;
typedef __attribute__((ext_vector_type(4)))  __bf16 v4bf;
typedef __attribute__((ext_vector_type(8)))  float  v8f;

union ABFrag { v16bf v; v8bf h[2]; };

// WMMA bf16: D = A(16x32) * B(32x16) + C(16x16 f32)
__device__ __forceinline__ v8f wmma_bf16(v16bf a, v16bf b, v8f c) {
  return __builtin_amdgcn_wmma_f32_16x16x32_bf16(false, a, false, b, (short)0, c,
                                                 false, false);
}

// A-fragment from LDS tile (row-major, KT bf16 per row).
// A layout (contiguous, matching B): lanes 0-15 hold K=0..15 (elems 0-15);
// lanes 16-31 hold K=16..31.
__device__ __forceinline__ v16bf load_a_lds(const bf16_t* lds, int row, int s32,
                                            int half) {
  ABFrag u;
  const bf16_t* p = lds + row * KT + s32 * 32;
  u.h[0] = *(const v8bf*)(p + half * 16);
  u.h[1] = *(const v8bf*)(p + half * 16 + 8);
  return u.v;
}

// Async global -> LDS stage of a 128 x KT bf16 tile (16 KB).
// 1024 chunks of 16B; 256 threads x 4 async b128 ops each (ASYNCcnt += 4/wave).
__device__ __forceinline__ void stage_tile_async(unsigned lds_base,
                                                 const bf16_t* gbase,
                                                 size_t row_stride, size_t k0,
                                                 int tid) {
#pragma unroll
  for (int i = 0; i < 4; ++i) {
    int q = tid + 256 * i;           // chunk id
    int row = q >> 3;                // 8 x 16B chunks per 128B row
    int c8 = q & 7;
    const bf16_t* g = gbase + (size_t)row * row_stride + k0 + (size_t)c8 * 8;
    unsigned l = lds_base + (unsigned)q * 16;
    asm volatile("global_load_async_to_lds_b128 %0, %1, off"
                 :: "v"(l), "v"(g) : "memory");
  }
}

__device__ __forceinline__ void wait_async_le4() {
  asm volatile("s_wait_asynccnt 0x4" ::: "memory");
}
__device__ __forceinline__ void wait_async_le0() {
  asm volatile("s_wait_asynccnt 0x0" ::: "memory");
}

// ---------------- fp32 -> bf16 convert ----------------
__global__ void f32_to_bf16_kernel(const float* __restrict__ src,
                                   bf16_t* __restrict__ dst, long n4) {
  long i = (long)blockIdx.x * blockDim.x + threadIdx.x;
  long stride = (long)gridDim.x * blockDim.x;
  for (; i < n4; i += stride) {
    float4 v = ((const float4*)src)[i];
    v4bf o;
    o.x = (__bf16)v.x; o.y = (__bf16)v.y; o.z = (__bf16)v.z; o.w = (__bf16)v.w;
    ((v4bf*)dst)[i] = o;
  }
}

// ---------------- router: logits, softmax, top-2 ----------------
__global__ __launch_bounds__(256) void router_kernel(
    const float* __restrict__ x, const float* __restrict__ Wr,
    int* __restrict__ topi, float* __restrict__ topw, float* __restrict__ probs) {
  const int wave = threadIdx.x >> 5, lane = threadIdx.x & 31;
  const int n = blockIdx.x * 8 + wave;
  float acc[NEXP];
#pragma unroll
  for (int e = 0; e < NEXP; ++e) acc[e] = 0.f;
  for (int d = lane; d < DMODEL; d += 32) {
    float xv = x[(size_t)n * DMODEL + d];
#pragma unroll
    for (int e = 0; e < NEXP; ++e) acc[e] += xv * Wr[e * DMODEL + d];
  }
#pragma unroll
  for (int off = 16; off > 0; off >>= 1) {
#pragma unroll
    for (int e = 0; e < NEXP; ++e) acc[e] += __shfl_xor(acc[e], off, 32);
  }
  if (lane == 0) {
    float m = acc[0];
#pragma unroll
    for (int e = 1; e < NEXP; ++e) m = fmaxf(m, acc[e]);
    float p[NEXP], s = 0.f;
#pragma unroll
    for (int e = 0; e < NEXP; ++e) { p[e] = __expf(acc[e] - m); s += p[e]; }
#pragma unroll
    for (int e = 0; e < NEXP; ++e) { p[e] /= s; probs[n * NEXP + e] = p[e]; }
    int i0 = 0; float v0 = p[0];
#pragma unroll
    for (int e = 1; e < NEXP; ++e) if (p[e] > v0) { v0 = p[e]; i0 = e; }
    int i1 = -1; float v1 = -1.f;
#pragma unroll
    for (int e = 0; e < NEXP; ++e)
      if (e != i0 && p[e] > v1) { v1 = p[e]; i1 = e; }
    float sw = v0 + v1 + 1e-9f;
    topi[n * TOPK + 0] = i0; topi[n * TOPK + 1] = i1;
    topw[n * TOPK + 0] = v0 / sw; topw[n * TOPK + 1] = v1 / sw;
  }
}

// ---------------- slot-major rank: histogram / scan / assign ----------------
__global__ void count_kernel(const int* __restrict__ topi, int* __restrict__ blk_counts) {
  __shared__ int cnt[NEXP];
  if (threadIdx.x < NEXP) cnt[threadIdx.x] = 0;
  __syncthreads();
  int a = blockIdx.x * 256 + threadIdx.x;
  int slot = a >> 13, n = a & (NTOK - 1);
  int e = topi[n * TOPK + slot];
  atomicAdd(&cnt[e], 1);
  __syncthreads();
  if (threadIdx.x < NEXP) blk_counts[blockIdx.x * NEXP + threadIdx.x] = cnt[threadIdx.x];
}

__global__ void scan_kernel(const int* __restrict__ blk_counts,
                            int* __restrict__ prefix, int* __restrict__ counts_e) {
  int e = threadIdx.x;
  if (e < NEXP) {
    int run = 0;
    for (int b = 0; b < NASSIGN / 256; ++b) {
      prefix[b * NEXP + e] = run;
      run += blk_counts[b * NEXP + e];
    }
    counts_e[e] = run;
  }
}

__global__ void fill_neg1_kernel(int* __restrict__ p, int n) {
  int i = blockIdx.x * 256 + threadIdx.x;
  if (i < n) p[i] = -1;
}

__global__ void assign_kernel(const int* __restrict__ topi, const int* __restrict__ prefix,
                              int* __restrict__ dest_arr, int* __restrict__ slot_token) {
  __shared__ int ex[256];
  int a = blockIdx.x * 256 + threadIdx.x;
  int slot = a >> 13, n = a & (NTOK - 1);
  int e = topi[n * TOPK + slot];
  ex[threadIdx.x] = e;
  __syncthreads();
  int lr = 0;
  for (int s = 0; s < threadIdx.x; ++s) lr += (ex[s] == e);
  int rank = prefix[blockIdx.x * NEXP + e] + lr;
  int dest = -1;
  if (rank < CAP) { dest = e * CAP + rank; slot_token[dest] = n; }
  dest_arr[a] = dest;
}

// ---------------- gather tokens into capacity buffer (bf16) ----------------
__global__ __launch_bounds__(256) void gather_kernel(
    const bf16_t* __restrict__ Xb, const int* __restrict__ slot_token,
    bf16_t* __restrict__ Xe) {
  int s = blockIdx.x;
  int tok = slot_token[s];
  int d = threadIdx.x * 4;  // 4 bf16 = 8 bytes
  uint2 z = make_uint2(0u, 0u);
  uint2 v = (tok >= 0) ? *(const uint2*)(Xb + (size_t)tok * DMODEL + d) : z;
  *(uint2*)(Xe + (size_t)s * DMODEL + d) = v;
}

// ---------------- GEMM1: H = silu(Xe*Wg^T) * (Xe*Wu^T), bf16 out ----------------
__global__ __launch_bounds__(256) void gemm_gateup_kernel(
    const bf16_t* __restrict__ Xe, const bf16_t* __restrict__ Wg,
    const bf16_t* __restrict__ Wu, bf16_t* __restrict__ Hb) {
  __shared__ bf16_t At[2][128 * KT];  // double-buffered 16 KB tiles
  const int e = blockIdx.z;
  const int m0 = blockIdx.y * 128;
  const int f0 = blockIdx.x * 128;
  const int tid = threadIdx.x;
  const int wave = tid >> 5, lane = tid & 31;
  const int wm = wave >> 1, wn = wave & 1;     // 4x2 wave grid: 32 rows x 64 cols
  const int half = lane >> 4, col = lane & 15;
  const size_t slotbase = (size_t)e * CAP + m0;
  const bf16_t* arow = Xe + slotbase * DMODEL;
  // B row base: weight row f = f0 + wn*64 + j*16 + col, contiguous over d
  const size_t wb = (((size_t)e * FFN) + f0 + wn * 64 + col) * DMODEL;
  const unsigned lds0 = (unsigned)(uintptr_t)(&At[0][0]);
  const unsigned lds1 = (unsigned)(uintptr_t)(&At[1][0]);

  v8f accG[2][4]; v8f accU[2][4];
#pragma unroll
  for (int s = 0; s < 2; ++s)
#pragma unroll
    for (int j = 0; j < 4; ++j) { accG[s][j] = (v8f)(0.f); accU[s][j] = (v8f)(0.f); }

  const int NKT = DMODEL / KT;  // 16 tiles
  stage_tile_async(lds0, arow, DMODEL, 0, tid);

  for (int kt = 0; kt < NKT; ++kt) {
    const int buf = kt & 1;
    if (kt + 1 < NKT) {
      stage_tile_async(buf ? lds0 : lds1, arow, DMODEL, (size_t)(kt + 1) * KT, tid);
      wait_async_le4();   // oldest 4 (current tile) have landed
    } else {
      wait_async_le0();
    }
    __syncthreads();      // all waves' stage of current tile visible

    const bf16_t* lt = &At[buf][0];
#pragma unroll
    for (int s32 = 0; s32 < 2; ++s32) {
      v16bf a0 = load_a_lds(lt, wm * 32 + col, s32, half);
      v16bf a1 = load_a_lds(lt, wm * 32 + 16 + col, s32, half);
      const size_t kd = (size_t)kt * KT + s32 * 32 + half * 16;
#pragma unroll
      for (int j = 0; j < 4; ++j) {
        v16bf bg = *(const v16bf*)(Wg + wb + (size_t)j * 16 * DMODEL + kd);
        v16bf bu = *(const v16bf*)(Wu + wb + (size_t)j * 16 * DMODEL + kd);
        accG[0][j] = wmma_bf16(a0, bg, accG[0][j]);
        accG[1][j] = wmma_bf16(a1, bg, accG[1][j]);
        accU[0][j] = wmma_bf16(a0, bu, accU[0][j]);
        accU[1][j] = wmma_bf16(a1, bu, accU[1][j]);
      }
    }
    __syncthreads();      // done reading buf before it is overwritten at kt+2
  }
  // epilogue: silu(gate)*up -> bf16
#pragma unroll
  for (int s = 0; s < 2; ++s) {
    size_t rb = slotbase + wm * 32 + s * 16 + half * 8;
#pragma unroll
    for (int j = 0; j < 4; ++j) {
      size_t cb = (size_t)f0 + wn * 64 + j * 16 + col;
#pragma unroll
      for (int r = 0; r < 8; ++r) {
        float g = accG[s][j][r], u = accU[s][j][r];
        float h = (g / (1.f + __expf(-g))) * u;
        Hb[(rb + r) * FFN + cb] = (__bf16)h;
      }
    }
  }
}

// ---------------- GEMM2: O = H * Wd^T, fp32 out ----------------
__global__ __launch_bounds__(256) void gemm_down_kernel(
    const bf16_t* __restrict__ Hb, const bf16_t* __restrict__ Wd,
    float* __restrict__ Ob) {
  __shared__ bf16_t At[2][128 * KT];
  const int e = blockIdx.z;
  const int m0 = blockIdx.y * 128;
  const int d0 = blockIdx.x * 128;
  const int tid = threadIdx.x;
  const int wave = tid >> 5, lane = tid & 31;
  const int wm = wave >> 1, wn = wave & 1;
  const int half = lane >> 4, col = lane & 15;
  const size_t slotbase = (size_t)e * CAP + m0;
  const bf16_t* arow = Hb + slotbase * FFN;
  const size_t wb = (((size_t)e * DMODEL) + d0 + wn * 64 + col) * FFN;
  const unsigned lds0 = (unsigned)(uintptr_t)(&At[0][0]);
  const unsigned lds1 = (unsigned)(uintptr_t)(&At[1][0]);

  v8f acc[2][4];
#pragma unroll
  for (int s = 0; s < 2; ++s)
#pragma unroll
    for (int j = 0; j < 4; ++j) acc[s][j] = (v8f)(0.f);

  const int NKT = FFN / KT;  // 64 tiles
  stage_tile_async(lds0, arow, FFN, 0, tid);

  for (int kt = 0; kt < NKT; ++kt) {
    const int buf = kt & 1;
    if (kt + 1 < NKT) {
      stage_tile_async(buf ? lds0 : lds1, arow, FFN, (size_t)(kt + 1) * KT, tid);
      wait_async_le4();
    } else {
      wait_async_le0();
    }
    __syncthreads();

    const bf16_t* lt = &At[buf][0];
#pragma unroll
    for (int s32 = 0; s32 < 2; ++s32) {
      v16bf a0 = load_a_lds(lt, wm * 32 + col, s32, half);
      v16bf a1 = load_a_lds(lt, wm * 32 + 16 + col, s32, half);
      const size_t kd = (size_t)kt * KT + s32 * 32 + half * 16;
#pragma unroll
      for (int j = 0; j < 4; ++j) {
        v16bf b = *(const v16bf*)(Wd + wb + (size_t)j * 16 * FFN + kd);
        acc[0][j] = wmma_bf16(a0, b, acc[0][j]);
        acc[1][j] = wmma_bf16(a1, b, acc[1][j]);
      }
    }
    __syncthreads();
  }
#pragma unroll
  for (int s = 0; s < 2; ++s) {
    size_t rb = slotbase + wm * 32 + s * 16 + half * 8;
#pragma unroll
    for (int j = 0; j < 4; ++j) {
      size_t cb = (size_t)d0 + wn * 64 + j * 16 + col;
#pragma unroll
      for (int r = 0; r < 8; ++r) Ob[(rb + r) * DMODEL + cb] = acc[s][j][r];
    }
  }
}

// ---------------- deterministic weighted combine ----------------
__global__ __launch_bounds__(256) void combine_kernel(
    const float* __restrict__ Ob, const int* __restrict__ dest_arr,
    const float* __restrict__ topw, float* __restrict__ out) {
  int n = blockIdx.x;
  int d = threadIdx.x * 4;
  float4 r = make_float4(0.f, 0.f, 0.f, 0.f);
#pragma unroll
  for (int k = 0; k < TOPK; ++k) {
    int dest = dest_arr[k * NTOK + n];
    if (dest >= 0) {
      float w = topw[n * TOPK + k];
      float4 v = *(const float4*)(Ob + (size_t)dest * DMODEL + d);
      r.x += w * v.x; r.y += w * v.y; r.z += w * v.z; r.w += w * v.w;
    }
  }
  *(float4*)(out + (size_t)n * DMODEL + d) = r;
}

// ---------------- aux loss (single block, deterministic) ----------------
__global__ __launch_bounds__(256) void aux_kernel(
    const float* __restrict__ probs, const int* __restrict__ counts_e,
    float* __restrict__ out_aux) {
  __shared__ float sp[256];
  int t = threadIdx.x;
  int e = t & 7, r0 = t >> 3;           // 32 partial groups per expert
  float s = 0.f;
  for (int n = r0; n < NTOK; n += 32) s += probs[n * NEXP + e];
  sp[t] = s;
  __syncthreads();
  for (int st = 128; st >= 8; st >>= 1) {
    if (t < st) sp[t] += sp[t + st];
    __syncthreads();
  }
  if (t == 0) {
    float c[NEXP], total = 0.f;
    for (int i = 0; i < NEXP; ++i) {
      int ci = counts_e[i]; if (ci > CAP) ci = CAP;
      c[i] = (float)ci; total += c[i];
    }
    total = fmaxf(total, 1.f);
    float acc = 0.f;
    for (int i = 0; i < NEXP; ++i)
      acc += (c[i] / total) * (sp[i] / (float)NTOK);
    *out_aux = 0.01f * (float)NEXP * acc;
  }
}

// ---------------- host launch ----------------
extern "C" void kernel_launch(void* const* d_in, const int* in_sizes, int n_in,
                              void* d_out, int out_size, void* d_ws, size_t ws_size,
                              hipStream_t stream) {
  const float* x  = (const float*)d_in[0];   // (1,8192,1024)
  const float* Wr = (const float*)d_in[1];   // (8,1024)
  const float* Wg = (const float*)d_in[2];   // (8,4096,1024)
  const float* Wu = (const float*)d_in[3];   // (8,4096,1024)
  const float* Wd = (const float*)d_in[4];   // (8,1024,4096)
  float* out = (float*)d_out;                // 8192*1024 floats + 1 aux

  // workspace carve-out (256B aligned)
  char* base = (char*)d_ws;
  size_t off = 0;
  auto carve = [&](size_t bytes) -> void* {
    void* p = base + off;
    off = (off + bytes + 255) & ~(size_t)255;
    return p;
  };
  int*    topi       = (int*)   carve((size_t)NTOK * TOPK * 4);
  float*  topw       = (float*) carve((size_t)NTOK * TOPK * 4);
  float*  probs      = (float*) carve((size_t)NTOK * NEXP * 4);
  int*    blk_counts = (int*)   carve((size_t)(NASSIGN / 256) * NEXP * 4);
  int*    prefix     = (int*)   carve((size_t)(NASSIGN / 256) * NEXP * 4);
  int*    counts_e   = (int*)   carve((size_t)NEXP * 4);
  int*    dest_arr   = (int*)   carve((size_t)NASSIGN * 4);
  int*    slot_token = (int*)   carve((size_t)NSLOT * 4);
  bf16_t* Xb         = (bf16_t*)carve((size_t)NTOK * DMODEL * 2);
  bf16_t* Wgb        = (bf16_t*)carve((size_t)NEXP * FFN * DMODEL * 2);
  bf16_t* Wub        = (bf16_t*)carve((size_t)NEXP * FFN * DMODEL * 2);
  bf16_t* Wdb        = (bf16_t*)carve((size_t)NEXP * DMODEL * FFN * 2);
  bf16_t* Xe         = (bf16_t*)carve((size_t)NSLOT * DMODEL * 2);
  bf16_t* Hb         = (bf16_t*)carve((size_t)NSLOT * FFN * 2);
  float*  Ob         = (float*) carve((size_t)NSLOT * DMODEL * 4);
  (void)ws_size; (void)n_in; (void)in_sizes; (void)out_size;

  // 1) fp32 -> bf16 conversions
  {
    long nx = (long)NTOK * DMODEL / 4;
    f32_to_bf16_kernel<<<4096, 256, 0, stream>>>(x, Xb, nx);
    long nw = (long)NEXP * FFN * DMODEL / 4;
    f32_to_bf16_kernel<<<8192, 256, 0, stream>>>(Wg, Wgb, nw);
    f32_to_bf16_kernel<<<8192, 256, 0, stream>>>(Wu, Wub, nw);
    f32_to_bf16_kernel<<<8192, 256, 0, stream>>>(Wd, Wdb, nw);
  }
  // 2) router
  router_kernel<<<NTOK / 8, 256, 0, stream>>>(x, Wr, topi, topw, probs);
  // 3) slot-major rank pipeline
  count_kernel<<<NASSIGN / 256, 256, 0, stream>>>(topi, blk_counts);
  scan_kernel<<<1, 32, 0, stream>>>(blk_counts, prefix, counts_e);
  fill_neg1_kernel<<<(NSLOT + 255) / 256, 256, 0, stream>>>(slot_token, NSLOT);
  assign_kernel<<<NASSIGN / 256, 256, 0, stream>>>(topi, prefix, dest_arr, slot_token);
  // 4) gather into capacity buffer
  gather_kernel<<<NSLOT, 256, 0, stream>>>(Xb, slot_token, Xe);
  // 5) fused gate/up + SiLU  (grid: F-tiles x M-tiles x experts)
  gemm_gateup_kernel<<<dim3(FFN / 128, CAP / 128, NEXP), 256, 0, stream>>>(Xe, Wgb, Wub, Hb);
  // 6) down projection
  gemm_down_kernel<<<dim3(DMODEL / 128, CAP / 128, NEXP), 256, 0, stream>>>(Hb, Wdb, Ob);
  // 7) weighted combine back to tokens
  combine_kernel<<<NTOK, 256, 0, stream>>>(Ob, dest_arr, topw, out);
  // 8) aux loss -> last output element
  aux_kernel<<<1, 256, 0, stream>>>(probs, counts_e, out + (size_t)NTOK * DMODEL);
}
